// SelfAttentionHead_35124242546876
// MI455X (gfx1250) — matrix operand order repaired and checked by reading, not compiled, not run on hardware
//
#include <hip/hip_runtime.h>

// ---- problem constants (B=2, T=2048, C=1024, H=16, HS=64) ----
#define BB 2
#define TT 2048
#define CC 1024
#define HH 16
#define HS 64
#define N3C (3 * CC)     // 3072
#define MROWS (BB * TT)  // 4096

typedef __attribute__((ext_vector_type(16))) __bf16 v16bf;
typedef __attribute__((ext_vector_type(8)))  float  v8f;

union AFrag {
    v16bf v;
    uint4 u[2];
    unsigned short s[16];
};

__device__ __forceinline__ unsigned short f2bf(float f) {
    union { float f; unsigned u; } c;
    c.f = f;
    unsigned r = c.u + 0x7FFFu + ((c.u >> 16) & 1u);  // round-to-nearest-even
    return (unsigned short)(r >> 16);
}

// ---------------- kernel 1: fp32 -> bf16 conversion of x and W ----------------
__global__ void cvt_bf16_kernel(const float* __restrict__ x, const float* __restrict__ W,
                                unsigned short* __restrict__ xb, unsigned short* __restrict__ wb,
                                int nx, int nw) {
    int i = blockIdx.x * blockDim.x + threadIdx.x;
    if (i < nx) {
        xb[i] = f2bf(x[i]);
    } else if (i < nx + nw) {
        wb[i - nx] = f2bf(W[i - nx]);
    }
}

// ---------------- kernel 2: QKV projection GEMM -------------------------------
// out[m,n] = sum_k x[m,k] * W[n,k] + bias[n]   (m in [0,4096), n in [0,3072))
// one wave -> 64x64 tile, bf16 WMMA 16x16x32, f32 accumulate.
// Epilogue: Q,K -> [B,H,T,HS]; V -> transposed [B,H,HS,T] so the attention
// kernel's V B-fragments are contiguous b128 loads.
__global__ void __launch_bounds__(256, 1)
qkv_gemm_kernel(const unsigned short* __restrict__ xb,
                const unsigned short* __restrict__ wb,
                const float* __restrict__ bias,
                unsigned short* __restrict__ Qb,
                unsigned short* __restrict__ Kb,
                unsigned short* __restrict__ Vt) {
    const int lane = threadIdx.x & 31;
    const int lo = lane & 15;
    const int hi = lane >> 4;
    const int wave = blockIdx.x * (blockDim.x >> 5) + (threadIdx.x >> 5);

    const int nTilesN = N3C / 64;  // 48
    const int tm = wave / nTilesN;
    const int tn = wave % nTilesN;
    const int mbase = tm * 64;
    const int nbase = tn * 64;

    v8f acc[4][4];
#pragma unroll
    for (int i = 0; i < 4; ++i)
#pragma unroll
        for (int j = 0; j < 4; ++j) acc[i][j] = (v8f){};

    const unsigned short* arow = xb + (mbase + lo) * CC + 8 * hi;
    const unsigned short* brow = wb + (nbase + lo) * CC + 8 * hi;

    for (int k0 = 0; k0 < CC; k0 += 32) {
        AFrag a[4];
#pragma unroll
        for (int mt = 0; mt < 4; ++mt) {
            // A layout: lane holds row (mbase+mt*16+lo), K runs [k0+8*hi .. +7] and [+16 .. +23]
            const unsigned short* p = arow + mt * 16 * CC + k0;
            a[mt].u[0] = *(const uint4*)(p);
            a[mt].u[1] = *(const uint4*)(p + 16);
        }
#pragma unroll
        for (int nt = 0; nt < 4; ++nt) {
            // B layout: lane holds column (nbase+nt*16+lo) = W row, K runs [k0+8*hi .. +7] and [+16 .. +23]
            const unsigned short* p = brow + nt * 16 * CC + k0;
            AFrag bf;
            bf.u[0] = *(const uint4*)(p);
            bf.u[1] = *(const uint4*)(p + 16);
#pragma unroll
            for (int mt = 0; mt < 4; ++mt)
                acc[mt][nt] = __builtin_amdgcn_wmma_f32_16x16x32_bf16(
                    false, a[mt].v, false, bf.v, (short)0, acc[mt][nt], false, false);
        }
    }

    // epilogue: + bias, f32 -> bf16, scatter.  s,h are wave-uniform per nt.
#pragma unroll
    for (int nt = 0; nt < 4; ++nt) {
        const int n = nbase + nt * 16 + lo;
        const int s = n >> 10;        // 0=q 1=k 2=v
        const int c = n & (CC - 1);
        const int h = c >> 6;
        const int d = c & (HS - 1);
        const float bv = bias[n];
        if (s < 2) {
            unsigned short* dst = (s == 0) ? Qb : Kb;
#pragma unroll
            for (int mt = 0; mt < 4; ++mt) {
#pragma unroll
                for (int r = 0; r < 8; ++r) {
                    const int m = mbase + mt * 16 + 8 * hi + r;
                    const int bb = m >> 11;
                    const int t = m & (TT - 1);
                    dst[(((bb * HH + h) * TT) + t) * HS + d] = f2bf(acc[mt][nt][r] + bv);
                }
            }
        } else {
            // V transposed: Vt[b][h][d][t]
#pragma unroll
            for (int mt = 0; mt < 4; ++mt) {
#pragma unroll
                for (int r = 0; r < 8; ++r) {
                    const int m = mbase + mt * 16 + 8 * hi + r;
                    const int bb = m >> 11;
                    const int t = m & (TT - 1);
                    Vt[(((bb * HH + h) * HS) + d) * TT + t] = f2bf(acc[mt][nt][r] + bv);
                }
            }
        }
    }
}

// ---------------- kernel 3: causal flash attention ----------------------------
// one wave handles 16 query rows for one (b,h). Computes S^T = K*Q^T so the
// softmaxed P tile lands directly in the WMMA A-operand layout for O += P*V.
// V is pre-transposed [B,H,HS,T], so its B-fragments are contiguous b128 loads.
__global__ void __launch_bounds__(256, 1)
attn_kernel(const unsigned short* __restrict__ Qb,
            const unsigned short* __restrict__ Kb,
            const unsigned short* __restrict__ Vt,
            float* __restrict__ out) {
    const int lane = threadIdx.x & 31;
    const int lo = lane & 15;
    const int hi = lane >> 4;
    const int wave = blockIdx.x * (blockDim.x >> 5) + (threadIdx.x >> 5);

    const int b = wave >> 11;              // H*(T/16) = 2048 waves per batch
    const int rem = wave & 2047;
    const int h = rem >> 7;                // T/16 = 128 q-tiles per head
    const int qt = rem & 127;
    const int qbase = qt * 16;

    const long base = (long)((b * HH + h) * TT) * HS;
    const unsigned short* Qp = Qb + base;
    const unsigned short* Kp = Kb + base;
    const unsigned short* Vp = Vt + base;  // same element count per head, [HS][T]

    // Q fragments (B-operand), fixed for the wave: d-halves 0..31, 32..63
    AFrag qf[2];
#pragma unroll
    for (int dh = 0; dh < 2; ++dh) {
        const unsigned short* p = Qp + (qbase + lo) * HS + dh * 32 + 8 * hi;
        qf[dh].u[0] = *(const uint4*)(p);
        qf[dh].u[1] = *(const uint4*)(p + 16);
    }

    v8f o[4];
#pragma unroll
    for (int dt = 0; dt < 4; ++dt) o[dt] = (v8f){};
    float mrun = -3.0e38f;
    float lrun = 0.0f;

    const int qi = qbase + lo;
    const int qmax = qbase + 15;

    for (int kb = 0; kb <= qmax; kb += 32) {
        // S^T tiles: rows k (16 each), cols q (16). Contract over d=64.
        v8f st0 = (v8f){}, st1 = (v8f){};
#pragma unroll
        for (int dh = 0; dh < 2; ++dh) {
            const int dk = dh * 32;
            AFrag a0, a1;
            const unsigned short* p0 = Kp + (kb + lo) * HS + dk + 8 * hi;
            a0.u[0] = *(const uint4*)(p0);
            a0.u[1] = *(const uint4*)(p0 + 16);
            const unsigned short* p1 = Kp + (kb + 16 + lo) * HS + dk + 8 * hi;
            a1.u[0] = *(const uint4*)(p1);
            a1.u[1] = *(const uint4*)(p1 + 16);
            st0 = __builtin_amdgcn_wmma_f32_16x16x32_bf16(false, a0.v, false, qf[dh].v,
                                                          (short)0, st0, false, false);
            st1 = __builtin_amdgcn_wmma_f32_16x16x32_bf16(false, a1.v, false, qf[dh].v,
                                                          (short)0, st1, false, false);
        }

        // scale + causal mask; track per-lane max over the 16 held scores
        float pm = -3.0e38f;
#pragma unroll
        for (int r = 0; r < 8; ++r) {
            const int kk = kb + 8 * hi + r;
            float v0 = st0[r] * 0.125f;
            float v1 = st1[r] * 0.125f;
            if (kk > qi) v0 = -3.0e38f;
            if (kk + 16 > qi) v1 = -3.0e38f;
            st0[r] = v0;
            st1[r] = v1;
            pm = fmaxf(pm, fmaxf(v0, v1));
        }
        pm = fmaxf(pm, __shfl_xor(pm, 16, 32));     // combine lane pair sharing column q
        const float mnew = fmaxf(mrun, pm);
        const float alpha = __expf(mrun - mnew);
        mrun = mnew;

        // exponentiate, build P fragment (A-operand layout), partial row-sums
        AFrag pf;
        float rs = 0.0f;
#pragma unroll
        for (int r = 0; r < 8; ++r) {
            const float p0 = __expf(st0[r] - mnew);
            const float p1 = __expf(st1[r] - mnew);
            rs += p0 + p1;
            pf.s[r] = f2bf(p0);
            pf.s[8 + r] = f2bf(p1);
        }
        rs += __shfl_xor(rs, 16, 32);
        lrun = lrun * alpha + rs;

        // rescale O accumulators (alpha lives in lane == q row index)
#pragma unroll
        for (int r = 0; r < 8; ++r) {
            const float ar = __shfl(alpha, 8 * hi + r, 32);
#pragma unroll
            for (int dt = 0; dt < 4; ++dt) o[dt][r] *= ar;
        }

        // O += P * V ; V B-fragment: lane = column d, t runs [kb+8*hi .. +7] and [+16 .. +23]
#pragma unroll
        for (int dt = 0; dt < 4; ++dt) {
            AFrag vf;
            const unsigned short* p = Vp + (dt * 16 + lo) * TT + kb + 8 * hi;
            vf.u[0] = *(const uint4*)(p);
            vf.u[1] = *(const uint4*)(p + 16);
            o[dt] = __builtin_amdgcn_wmma_f32_16x16x32_bf16(false, pf.v, false, vf.v,
                                                            (short)0, o[dt], false, false);
        }
    }

    // normalize and write out [B,T,C] fp32
#pragma unroll
    for (int r = 0; r < 8; ++r) {
        const float li = __shfl(lrun, 8 * hi + r, 32);
        const float inv = 1.0f / li;
        const int t = qbase + 8 * hi + r;
        const long obase = ((long)(b * TT + t) << 10) + h * HS;
#pragma unroll
        for (int dt = 0; dt < 4; ++dt)
            out[obase + dt * 16 + lo] = o[dt][r] * inv;
    }
}

// ---------------------------------------------------------------------------
extern "C" void kernel_launch(void* const* d_in, const int* in_sizes, int n_in,
                              void* d_out, int out_size, void* d_ws, size_t ws_size,
                              hipStream_t stream) {
    const float* x = (const float*)d_in[0];   // [B,T,C]
    const float* W = (const float*)d_in[1];   // [3C,C]
    const float* bias = (const float*)d_in[2];// [3C]
    float* out = (float*)d_out;               // [B,T,C]

    const int nx = MROWS * CC;                // 4,194,304
    const int nw = N3C * CC;                  // 3,145,728
    const int nqkv = BB * HH * TT * HS;       // 4,194,304 per buffer

    unsigned short* xb = (unsigned short*)d_ws;
    unsigned short* wb = xb + nx;
    unsigned short* Qb = wb + nw;
    unsigned short* Kb = Qb + nqkv;
    unsigned short* Vt = Kb + nqkv;

    // 1) convert x, W to bf16
    {
        const int total = nx + nw;
        const int threads = 256;
        const int blocks = (total + threads - 1) / threads;
        cvt_bf16_kernel<<<blocks, threads, 0, stream>>>(x, W, xb, wb, nx, nw);
    }

    // 2) QKV GEMM: (4096/64)*(3072/64) = 3072 wave-tiles, 8 waves/block
    {
        const int tiles = (MROWS / 64) * (N3C / 64);  // 3072
        qkv_gemm_kernel<<<tiles / 8, 256, 0, stream>>>(xb, wb, bias, Qb, Kb, Vt);
    }

    // 3) attention: B*H*(T/16) = 4096 waves, 8 waves/block
    {
        const int waves = BB * HH * (TT / 16);        // 4096
        attn_kernel<<<waves / 8, 256, 0, stream>>>(Qb, Kb, Vt, out);
    }

    (void)in_sizes; (void)n_in; (void)out_size; (void)ws_size;
}
